// SCAM_4518305595784
// MI455X (gfx1250) — matrix operand order repaired and checked by reading, not compiled, not run on hardware
//
#include <hip/hip_runtime.h>
#include <hip/hip_bf16.h>
#include <math.h>

// ---------------- types for gfx1250 WMMA ----------------
typedef __attribute__((ext_vector_type(16))) __bf16 v16bf;
typedef __attribute__((ext_vector_type(8)))  float  v8f;

union Frag16 { uint4 u[2]; v16bf v; };

#define C64 64
#define HW 256
#define PLANE 65536   // 256*256
#define NWIN_PER_BLK 8

// ---------------- async LDS path (guarded: falls back if builtins absent) ----------------
#if defined(__HIP_DEVICE_COMPILE__) && \
    __has_builtin(__builtin_amdgcn_global_load_async_to_lds_b32) && \
    __has_builtin(__builtin_amdgcn_global_store_async_from_lds_b32)
#define USE_ASYNC_LDS 1
#define AS1 __attribute__((address_space(1)))
#define AS3 __attribute__((address_space(3)))
#else
#define USE_ASYNC_LDS 0
#endif

__device__ inline void wait_asynccnt0() {
#if defined(__HIP_DEVICE_COMPILE__)
#if __has_builtin(__builtin_amdgcn_s_wait_asynccnt)
    __builtin_amdgcn_s_wait_asynccnt(0);
#else
    asm volatile("s_wait_asynccnt 0x0" ::: "memory");
#endif
#endif
}

// ---------------- fragment loaders (contiguous k = 16*half + e in A and B alike) ----------------
// A matrix 16x32 bf16, row-major source with leading dim lda (elements).
__device__ inline v16bf ldA_lds(const __bf16* A, int lda, int m0, int k0, int lane) {
    int m = m0 + (lane & 15);
    int half = lane >> 4;
    const __bf16* p = A + m * lda + k0 + half * 16;
    Frag16 f;
    f.u[0] = *(const uint4*)(p);
    f.u[1] = *(const uint4*)(p + 8);
    return f.v;
}

// B matrix 32x16 bf16 where B[k][n] = W[n][k]  (W row-major [O][I], ldw elems)
__device__ inline v16bf ldB_w(const __bf16* W, int ldw, int k0, int n0, int lane) {
    int n = n0 + (lane & 15);
    int half = lane >> 4;
    const uint4* p = (const uint4*)(W + n * ldw + k0 + half * 16);
    Frag16 f;
    f.u[0] = p[0];
    f.u[1] = p[1];
    return f.v;
}

// B matrix 32x16 bf16, plain row-major B[k][n], leading dim ldb
__device__ inline v16bf ldB_rm(const __bf16* B, int ldb, int k0, int n0, int lane) {
    int n = n0 + (lane & 15);
    int half = lane >> 4;
    v16bf f;
#pragma unroll
    for (int e = 0; e < 16; e++) f[e] = B[(k0 + half * 16 + e) * ldb + n];
    return f;
}

// A matrix from f32 halo tile [10][10][64] with conv shift (ky,kx), cvt to bf16
__device__ inline v16bf ldA_shift(const float* XH, int m0, int k0, int ky, int kx, int lane) {
    int m = m0 + (lane & 15);
    int half = lane >> 4;
    int ty = m >> 3, tx = m & 7;
    const float* p = XH + ((ty + ky) * 10 + (tx + kx)) * 64 + k0 + half * 16;
    v16bf f;
#pragma unroll
    for (int e = 0; e < 8; e++) f[e] = (__bf16)p[e];
#pragma unroll
    for (int e = 0; e < 8; e++) f[8 + e] = (__bf16)p[8 + e];
    return f;
}

// B for 3x3 conv: B[k=cin][n=cout] = ms1w[((n)*64 + k)*9 + koff]
__device__ inline v16bf ldB_conv(const __bf16* W, int k0, int n0, int koff, int lane) {
    int n = n0 + (lane & 15);
    int half = lane >> 4;
    v16bf f;
#pragma unroll
    for (int e = 0; e < 16; e++) f[e] = W[(n * 64 + k0 + half * 16 + e) * 9 + koff];
    return f;
}

__device__ inline v8f wmma_bf(v16bf a, v16bf b, v8f c) {
    return __builtin_amdgcn_wmma_f32_16x16x32_bf16(false, a, false, b, (short)0, c, false, false);
}

__device__ inline float gelu_tanh(float v) {
    const float k0 = 0.7978845608028654f, k1 = 0.044715f;
    return 0.5f * v * (1.0f + tanhf(k0 * (v + k1 * v * v * v)));
}

// ---------------- small prep kernels ----------------
__global__ __launch_bounds__(256) void cvt_bf16_k(const float* __restrict__ src,
                                                  __bf16* __restrict__ dst, int n) {
    int i = blockIdx.x * 256 + threadIdx.x;
    if (i < n) dst[i] = (__bf16)src[i];
}

__global__ __launch_bounds__(256) void gap_k(const float* __restrict__ x, float* __restrict__ gap) {
    int bc = blockIdx.x;  // 0..511  (b*64+c)
    const float* p = x + (size_t)bc * PLANE;
    float s = 0.f;
    for (int i = threadIdx.x; i < PLANE; i += 256) s += p[i];
    __shared__ float red[256];
    red[threadIdx.x] = s;
    __syncthreads();
    for (int st = 128; st > 0; st >>= 1) {
        if (threadIdx.x < st) red[threadIdx.x] += red[threadIdx.x + st];
        __syncthreads();
    }
    if (threadIdx.x == 0) gap[bc] = red[0] * (1.0f / (float)PLANE);
}

__global__ __launch_bounds__(64) void se_k(const float* __restrict__ gap,
                                           const float* __restrict__ se1w, const float* __restrict__ se1b,
                                           const float* __restrict__ se2w, const float* __restrict__ se2b,
                                           float* __restrict__ gate) {
    __shared__ float h[4];
    int t = threadIdx.x;
    for (int b = 0; b < 8; b++) {
        if (t < 4) {
            float s = se1b[t];
            for (int c = 0; c < 64; c++) s += gap[b * 64 + c] * se1w[t * 64 + c];
            h[t] = fmaxf(s, 0.f);
        }
        __syncthreads();
        float s = se2b[t];
        for (int j = 0; j < 4; j++) s += h[j] * se2w[t * 4 + j];
        gate[b * 64 + t] = 1.0f / (1.0f + expf(-s));
        __syncthreads();
    }
}

// ---------------- fused window-block kernel ----------------
__global__ __launch_bounds__(256) void swin_block_k(
    const float* __restrict__ x,
    const __bf16* __restrict__ qkw, const __bf16* __restrict__ vw,
    const __bf16* __restrict__ outw, const __bf16* __restrict__ ms1w,
    const __bf16* __restrict__ ms2w, const __bf16* __restrict__ fc1w,
    const __bf16* __restrict__ fc2w,
    const float* __restrict__ qk_b, const float* __restrict__ v_b,
    const float* __restrict__ out_b, const float* __restrict__ ms1_b,
    const float* __restrict__ ms2_b, const float* __restrict__ fc1_b,
    const float* __restrict__ fc2_b,
    const float* __restrict__ ln_g, const float* __restrict__ ln_b,
    const float* __restrict__ gate,
    float* __restrict__ outp) {
    extern __shared__ char smem[];
    float*  XH = (float*)smem;                    // [10*10][64] f32 halo   25600 B
    __bf16* XB = (__bf16*)(smem + 25600);         // x tokens / P / tn       8192 B
    __bf16* QS = (__bf16*)(smem + 33792);         // Q / H                   8192 B
    __bf16* KS = (__bf16*)(smem + 41984);         // K / O                   8192 B
    __bf16* VS = (__bf16*)(smem + 50176);         // V / ms_mid              8192 B
    float*  SF = (float*)(smem + 58368);          // S logits / x1 f32      16384 B
                                                  // total                  74752 B
    const int t = threadIdx.x;
    const int lane = t & 31;
    const int wave = t >> 5;  // 0..7

    for (int wi = 0; wi < NWIN_PER_BLK; wi++) {
        int wid = blockIdx.x * NWIN_PER_BLK + wi;  // 0..8191
        int b = wid >> 10;
        int rem = wid & 1023;
        int wy = rem >> 5, wx = rem & 31;
        int y0 = wy * 8, x0 = wx * 8;
        const float* xb = x + (size_t)b * C64 * PLANE;

        // ---- load 10x10x64 halo tile (zero-padded), token-major [p][c] ----
#if USE_ASYNC_LDS
        bool interior = (y0 >= 8) && (y0 <= 240) && (x0 >= 8) && (x0 <= 240);
        if (interior) {
            // async DMA: global -> LDS, no VGPR round trip (ASYNCcnt tracked)
#pragma unroll 5
            for (int k2 = 0; k2 < 25; k2++) {
                int f = k2 * 256 + t;  // 0..6399
                int c = f / 100, p = f % 100;
                const float* gp = xb + (size_t)c * PLANE + (y0 - 1 + p / 10) * 256 + (x0 - 1 + p % 10);
                __builtin_amdgcn_global_load_async_to_lds_b32(
                    (AS1 int*)gp, (AS3 int*)(XH + p * 64 + c), 0, 0);
            }
        } else
#endif
        {
#pragma unroll 5
            for (int k2 = 0; k2 < 25; k2++) {
                int f = k2 * 256 + t;  // 0..6399
                int c = f / 100, p = f % 100;
                int gy = y0 - 1 + p / 10;
                int gx = x0 - 1 + p % 10;
                float v = 0.f;
                if ((unsigned)gy < 256u && (unsigned)gx < 256u) {
                    const float* gp = xb + (size_t)c * PLANE + gy * 256 + gx;
                    v = *gp;
                    if (gx + 8 < 256) __builtin_prefetch(gp + 8, 0, 0);  // next window
                }
                XH[p * 64 + c] = v;
            }
        }
        // covers this window's halo async-loads and the previous window's
        // async result stores (SF is not overwritten until after next barrier)
        wait_asynccnt0();
        __syncthreads();

        // ---- center tokens to bf16 [64 tok][64 c] ----
#pragma unroll 4
        for (int k2 = 0; k2 < 16; k2++) {
            int f = k2 * 256 + t;
            int tok = f >> 6, c = f & 63;
            XB[f] = (__bf16)XH[(((tok >> 3) + 1) * 10 + (tok & 7) + 1) * 64 + c];
        }
        __syncthreads();

        // ---- QKV projections: 16 tiles each, 2 tiles/wave ----
#pragma unroll
        for (int r = 0; r < 2; r++) {
            int tile = wave * 2 + r;
            int m0 = (tile >> 2) * 16, n0 = (tile & 3) * 16;
            v8f aq = {}, ak = {}, av = {};
#pragma unroll
            for (int k0 = 0; k0 < 64; k0 += 32) {
                v16bf a = ldA_lds(XB, 64, m0, k0, lane);
                aq = wmma_bf(a, ldB_w(qkw, 64, k0, n0, lane), aq);
                ak = wmma_bf(a, ldB_w(qkw + 64 * 64, 64, k0, n0, lane), ak);
                av = wmma_bf(a, ldB_w(vw, 64, k0, n0, lane), av);
            }
            int n = n0 + (lane & 15), half = lane >> 4;
            float bq = qk_b[n], bk = qk_b[64 + n], bv = v_b[n];
#pragma unroll
            for (int j = 0; j < 8; j++) {
                int m = m0 + j + half * 8;
                QS[m * 64 + n] = (__bf16)(aq[j] + bq);
                KS[m * 64 + n] = (__bf16)(ak[j] + bk);
                VS[m * 64 + n] = (__bf16)(av[j] + bv);
            }
        }
        __syncthreads();

        // ---- S = Q K^T * scale ----
#pragma unroll
        for (int r = 0; r < 2; r++) {
            int tile = wave * 2 + r;
            int m0 = (tile >> 2) * 16, n0 = (tile & 3) * 16;
            v8f s = {};
#pragma unroll
            for (int k0 = 0; k0 < 64; k0 += 32)
                s = wmma_bf(ldA_lds(QS, 64, m0, k0, lane), ldB_w(KS, 64, k0, n0, lane), s);
            int n = n0 + (lane & 15), half = lane >> 4;
#pragma unroll
            for (int j = 0; j < 8; j++) SF[(m0 + j + half * 8) * 64 + n] = s[j] * 0.125f;
        }
        __syncthreads();

        // ---- softmax rows -> P (bf16 into XB) ----
        if (t < 64) {
            float mx = -1e30f;
            for (int s = 0; s < 64; s++) mx = fmaxf(mx, SF[t * 64 + s]);
            float sum = 0.f;
            for (int s = 0; s < 64; s++) sum += __expf(SF[t * 64 + s] - mx);
            float inv = 1.0f / sum;
            for (int s = 0; s < 64; s++)
                XB[t * 64 + s] = (__bf16)(__expf(SF[t * 64 + s] - mx) * inv);
        }
        __syncthreads();

        // ---- O = P V  (bf16 into KS) ----
#pragma unroll
        for (int r = 0; r < 2; r++) {
            int tile = wave * 2 + r;
            int m0 = (tile >> 2) * 16, n0 = (tile & 3) * 16;
            v8f o = {};
#pragma unroll
            for (int k0 = 0; k0 < 64; k0 += 32)
                o = wmma_bf(ldA_lds(XB, 64, m0, k0, lane), ldB_rm(VS, 64, k0, n0, lane), o);
            int n = n0 + (lane & 15), half = lane >> 4;
#pragma unroll
            for (int j = 0; j < 8; j++) KS[(m0 + j + half * 8) * 64 + n] = (__bf16)o[j];
        }
        __syncthreads();

        // ---- 3x3 conv as 9 shifted GEMMs: ms_mid[64][32] relu -> VS ----
        {
            int m0 = (wave >> 1) * 16, n0 = (wave & 1) * 16;
            v8f acc = {};
#pragma unroll
            for (int ky = 0; ky < 3; ky++)
#pragma unroll
                for (int kx = 0; kx < 3; kx++) {
                    int koff = ky * 3 + kx;
#pragma unroll
                    for (int k0 = 0; k0 < 64; k0 += 32)
                        acc = wmma_bf(ldA_shift(XH, m0, k0, ky, kx, lane),
                                      ldB_conv(ms1w, k0, n0, koff, lane), acc);
                }
            int n = n0 + (lane & 15), half = lane >> 4;
            float bb = ms1_b[n];
#pragma unroll
            for (int j = 0; j < 8; j++)
                VS[(m0 + j + half * 8) * 32 + n] = (__bf16)fmaxf(acc[j] + bb, 0.f);
        }
        __syncthreads();

        // ---- x1 = x + ms2(ms_mid) ; then += (O@outw + b)*gate  (both own-wave tiles) ----
#pragma unroll
        for (int r = 0; r < 2; r++) {
            int tile = wave * 2 + r;
            int m0 = (tile >> 2) * 16, n0 = (tile & 3) * 16;
            v8f acc = {};
            acc = wmma_bf(ldA_lds(VS, 32, m0, 0, lane), ldB_w(ms2w, 32, 0, n0, lane), acc);
            int n = n0 + (lane & 15), half = lane >> 4;
            float bb = ms2_b[n];
#pragma unroll
            for (int j = 0; j < 8; j++) {
                int m = m0 + j + half * 8;
                int ty = m >> 3, tx = m & 7;
                SF[m * 64 + n] = XH[((ty + 1) * 10 + tx + 1) * 64 + n] + acc[j] + bb;
            }
            v8f ao = {};
#pragma unroll
            for (int k0 = 0; k0 < 64; k0 += 32)
                ao = wmma_bf(ldA_lds(KS, 64, m0, k0, lane), ldB_w(outw, 64, k0, n0, lane), ao);
            float bo = out_b[n], g = gate[b * 64 + n];
#pragma unroll
            for (int j = 0; j < 8; j++) SF[(m0 + j + half * 8) * 64 + n] += (ao[j] + bo) * g;
        }
        __syncthreads();

        // ---- LayerNorm over channels -> tn (bf16 into XB) ----
        if (t < 64) {
            float mu = 0.f;
            for (int c = 0; c < 64; c++) mu += SF[t * 64 + c];
            mu *= (1.0f / 64.0f);
            float var = 0.f;
            for (int c = 0; c < 64; c++) {
                float d = SF[t * 64 + c] - mu;
                var += d * d;
            }
            float rs = rsqrtf(var * (1.0f / 64.0f) + 1e-5f);
            for (int c = 0; c < 64; c++)
                XB[t * 64 + c] = (__bf16)((SF[t * 64 + c] - mu) * rs * ln_g[c] + ln_b[c]);
        }
        __syncthreads();

        // ---- H = gelu(tn @ fc1^T + b) -> QS ----
#pragma unroll
        for (int r = 0; r < 2; r++) {
            int tile = wave * 2 + r;
            int m0 = (tile >> 2) * 16, n0 = (tile & 3) * 16;
            v8f acc = {};
#pragma unroll
            for (int k0 = 0; k0 < 64; k0 += 32)
                acc = wmma_bf(ldA_lds(XB, 64, m0, k0, lane), ldB_w(fc1w, 64, k0, n0, lane), acc);
            int n = n0 + (lane & 15), half = lane >> 4;
            float bb = fc1_b[n];
#pragma unroll
            for (int j = 0; j < 8; j++)
                QS[(m0 + j + half * 8) * 64 + n] = (__bf16)gelu_tanh(acc[j] + bb);
        }
        __syncthreads();

        // ---- out tile = x1 + H @ fc2^T + b  (accumulate into SF) ----
#pragma unroll
        for (int r = 0; r < 2; r++) {
            int tile = wave * 2 + r;
            int m0 = (tile >> 2) * 16, n0 = (tile & 3) * 16;
            v8f acc = {};
#pragma unroll
            for (int k0 = 0; k0 < 64; k0 += 32)
                acc = wmma_bf(ldA_lds(QS, 64, m0, k0, lane), ldB_w(fc2w, 64, k0, n0, lane), acc);
            int n = n0 + (lane & 15), half = lane >> 4;
            float bb = fc2_b[n];
#pragma unroll
            for (int j = 0; j < 8; j++) SF[(m0 + j + half * 8) * 64 + n] += acc[j] + bb;
        }
        __syncthreads();

        // ---- write-out: LDS -> global directly via async store when available ----
#if USE_ASYNC_LDS
#pragma unroll 4
        for (int k2 = 0; k2 < 16; k2++) {
            int f = k2 * 256 + t;
            int c = f >> 6, tok = f & 63;
            float* gp = outp + (size_t)(b * 64 + c) * PLANE + (y0 + (tok >> 3)) * 256 + (x0 + (tok & 7));
            __builtin_amdgcn_global_store_async_from_lds_b32(
                (AS1 int*)gp, (AS3 int*)(SF + tok * 64 + c), 0, 0);
        }
        // completion enforced by wait_asynccnt0() at top of next window
        // (SF is only rewritten after that point); s_endpgm waits-idle at kernel end.
#else
#pragma unroll 4
        for (int k2 = 0; k2 < 16; k2++) {
            int f = k2 * 256 + t;
            int c = f >> 6, tok = f & 63;
            int ty = tok >> 3, tx = tok & 7;
            outp[(size_t)(b * 64 + c) * PLANE + (y0 + ty) * 256 + (x0 + tx)] = SF[tok * 64 + c];
        }
#endif
        __syncthreads();
    }
}

// ---------------- host launcher ----------------
extern "C" void kernel_launch(void* const* d_in, const int* in_sizes, int n_in,
                              void* d_out, int out_size, void* d_ws, size_t ws_size,
                              hipStream_t stream) {
    (void)in_sizes; (void)n_in; (void)out_size; (void)ws_size;
    const float* x     = (const float*)d_in[0];
    const float* qk_w  = (const float*)d_in[1];
    const float* qk_b  = (const float*)d_in[2];
    const float* v_w   = (const float*)d_in[3];
    const float* v_b   = (const float*)d_in[4];
    const float* out_w = (const float*)d_in[5];
    const float* out_b = (const float*)d_in[6];
    const float* ms1_w = (const float*)d_in[7];
    const float* ms1_b = (const float*)d_in[8];
    const float* ms2_w = (const float*)d_in[9];
    const float* ms2_b = (const float*)d_in[10];
    const float* se1_w = (const float*)d_in[11];
    const float* se1_b = (const float*)d_in[12];
    const float* se2_w = (const float*)d_in[13];
    const float* se2_b = (const float*)d_in[14];
    const float* ln_g  = (const float*)d_in[15];
    const float* ln_b  = (const float*)d_in[16];
    const float* fc1_w = (const float*)d_in[17];
    const float* fc1_b = (const float*)d_in[18];
    const float* fc2_w = (const float*)d_in[19];
    const float* fc2_b = (const float*)d_in[20];
    float* outp = (float*)d_out;

    // workspace layout: bf16 weights then gap/gate f32
    __bf16* wbf = (__bf16*)d_ws;
    __bf16* qkw  = wbf + 0;      // 8192
    __bf16* vw   = wbf + 8192;   // 4096
    __bf16* outw = wbf + 12288;  // 4096
    __bf16* ms1w = wbf + 16384;  // 18432
    __bf16* ms2w = wbf + 34816;  // 2048
    __bf16* fc1w = wbf + 36864;  // 4096
    __bf16* fc2w = wbf + 40960;  // 4096
    float* gap  = (float*)((char*)d_ws + 45056 * 2);  // 512 floats
    float* gate = gap + 512;                          // 512 floats

    cvt_bf16_k<<<(8192  + 255) / 256, 256, 0, stream>>>(qk_w,  qkw,  8192);
    cvt_bf16_k<<<(4096  + 255) / 256, 256, 0, stream>>>(v_w,   vw,   4096);
    cvt_bf16_k<<<(4096  + 255) / 256, 256, 0, stream>>>(out_w, outw, 4096);
    cvt_bf16_k<<<(18432 + 255) / 256, 256, 0, stream>>>(ms1_w, ms1w, 18432);
    cvt_bf16_k<<<(2048  + 255) / 256, 256, 0, stream>>>(ms2_w, ms2w, 2048);
    cvt_bf16_k<<<(4096  + 255) / 256, 256, 0, stream>>>(fc1_w, fc1w, 4096);
    cvt_bf16_k<<<(4096  + 255) / 256, 256, 0, stream>>>(fc2_w, fc2w, 4096);

    gap_k<<<512, 256, 0, stream>>>(x, gap);
    se_k<<<1, 64, 0, stream>>>(gap, se1_w, se1_b, se2_w, se2_b, gate);

    swin_block_k<<<1024, 256, 74752, stream>>>(
        x, qkw, vw, outw, ms1w, ms2w, fc1w, fc2w,
        qk_b, v_b, out_b, ms1_b, ms2_b, fc1_b, fc2_b,
        ln_g, ln_b, gate, outp);
}
